// GPTModel_57801669869655
// MI455X (gfx1250) — matrix operand order repaired and checked by reading, not compiled, not run on hardware
//
#include <hip/hip_runtime.h>

// ---------------------------------------------------------------------------
// GPT-2 small forward for MI455X (gfx1250, wave32, WMMA).
// All GEMMs run through v_wmma_f32_16x16x32_bf16 (fp32 in -> bf16 A/B, fp32 acc).
// Invariants exploited (all launches below respect them):
//   - M is a multiple of 128, K is a multiple of 32, A rows 16B-aligned.
//   - Only N can be ragged -> compile-time GUARDN variants.
// ---------------------------------------------------------------------------

typedef __attribute__((ext_vector_type(16))) __bf16   v16bf;
typedef __attribute__((ext_vector_type(8)))  float    v8f;
typedef unsigned long long u64_t;
typedef __attribute__((ext_vector_type(4)))  u64_t    v4u64;

__device__ __forceinline__ unsigned short f2bf(float f) {
  unsigned u = __builtin_bit_cast(unsigned, f);
  unsigned r = u + 0x7fffu + ((u >> 16) & 1u);   // round-to-nearest-even
  return (unsigned short)(r >> 16);
}

__device__ __forceinline__ u64_t pack4bf(float a, float b, float c, float d) {
  return (u64_t)f2bf(a) | ((u64_t)f2bf(b) << 16) |
         ((u64_t)f2bf(c) << 32) | ((u64_t)f2bf(d) << 48);
}

__device__ __forceinline__ float gelu_tanh_dev(float x) {
  const float c = 0.7978845608028654f;           // sqrt(2/pi)
  float x3 = x * x * x;
  return 0.5f * x * (1.0f + tanhf(c * (x + 0.044715f * x3)));
}

// ---------------------------------------------------------------------------
// Embedding: h[b,s,:] = tok_emb[x[b,s]] + pos_emb[s]
// ---------------------------------------------------------------------------
__global__ __launch_bounds__(256)
void embed_kernel(const int* __restrict__ x, const float* __restrict__ tok,
                  const float* __restrict__ pos, float* __restrict__ h,
                  int S, int D, long total) {
  long idx = (long)blockIdx.x * 256 + threadIdx.x;
  if (idx >= total) return;
  long row = idx / D;
  int  d   = (int)(idx - row * D);
  int  s   = (int)(row % S);
  int  t   = x[row];
  h[idx] = tok[(long)t * D + d] + pos[(long)s * D + d];
}

// ---------------------------------------------------------------------------
// LayerNorm (one 256-thread block per row of length D)
// ---------------------------------------------------------------------------
__global__ __launch_bounds__(256)
void ln_kernel(const float* __restrict__ x, const float* __restrict__ scale,
               const float* __restrict__ shift, float* __restrict__ y, int D) {
  __shared__ float red[256];
  const int tid = threadIdx.x;
  const float* xr = x + (long)blockIdx.x * D;
  float*       yr = y + (long)blockIdx.x * D;

  float s = 0.f;
  for (int i = tid; i < D; i += 256) s += xr[i];
  red[tid] = s; __syncthreads();
  for (int off = 128; off > 0; off >>= 1) {
    if (tid < off) red[tid] += red[tid + off];
    __syncthreads();
  }
  const float mean = red[0] / (float)D;
  __syncthreads();

  float s2 = 0.f;
  for (int i = tid; i < D; i += 256) { float d = xr[i] - mean; s2 += d * d; }
  red[tid] = s2; __syncthreads();
  for (int off = 128; off > 0; off >>= 1) {
    if (tid < off) red[tid] += red[tid + off];
    __syncthreads();
  }
  const float inv = rsqrtf(red[0] / (float)D + 1e-5f);

  for (int i = tid; i < D; i += 256)
    yr[i] = scale[i] * (xr[i] - mean) * inv + shift[i];
}

// ---------------------------------------------------------------------------
// Causal softmax over scores rows. grid.x = B*H*S, row length S.
// ---------------------------------------------------------------------------
__global__ __launch_bounds__(256)
void softmax_causal(float* sc, int S, float scale) {
  __shared__ float red[256];
  const int tid = threadIdx.x;
  const long row = blockIdx.x;
  const int  i   = (int)(row % S);
  float* sr = sc + row * (long)S;

  float m = -3.0e38f;
  for (int j = tid; j <= i; j += 256) m = fmaxf(m, sr[j] * scale);
  red[tid] = m; __syncthreads();
  for (int off = 128; off > 0; off >>= 1) {
    if (tid < off) red[tid] = fmaxf(red[tid], red[tid + off]);
    __syncthreads();
  }
  const float mx = red[0];
  __syncthreads();

  float s = 0.f;
  for (int j = tid; j <= i; j += 256) {
    float e = __expf(sr[j] * scale - mx);
    sr[j] = e; s += e;
  }
  red[tid] = s; __syncthreads();
  for (int off = 128; off > 0; off >>= 1) {
    if (tid < off) red[tid] += red[tid + off];
    __syncthreads();
  }
  const float inv = 1.0f / red[0];
  for (int j = tid; j < S; j += 256) sr[j] = (j <= i) ? sr[j] * inv : 0.f;
}

// ---------------------------------------------------------------------------
// Generic bf16-WMMA GEMM:  C[M,N] = epilogue( A[M,K] @ W[K,N] )
//   256 threads = 8 wave32; block tile 128x128; wave tile 32x64 (2x4 WMMA).
//   Template: TRANSB (W stored [N,K]), GELU, GUARDN (ragged N), ALW (ldw%4==0).
// Fragment layouts per CDNA5 ISA 7.12.2:
//   A (16x32 bf16): lane L holds row M=L%16; element e -> K=(e<8?0:16)+(L/16)*8+(e%8)
//   B (32x16 bf16): lane L holds col N=L%16; element e -> K=(e<8?0:16)+(L/16)*8+(e%8)
// ---------------------------------------------------------------------------
template <bool TRANSB, bool GELU, bool GUARDN, bool ALW>
__global__ __launch_bounds__(256)
void gemm_bf16_wmma(const float* __restrict__ A, int lda, long sA1, long sA2,
                    const float* __restrict__ W, int ldw, long sW1, long sW2,
                    float* C, int ldc, long sC1, long sC2,
                    const float* R, const float* __restrict__ bias,
                    int M, int N, int K, int hdiv) {
  __shared__ unsigned short lA[128][36];   // 128x32 tile (stride 36 -> 8B rows)
  __shared__ unsigned short lB[32][132];   // 32x128 tile (stride 132 -> 8B rows)

  const int zb = blockIdx.z / hdiv;
  const int zh = blockIdx.z % hdiv;
  A += (long)zb * sA1 + (long)zh * sA2;
  W += (long)zb * sW1 + (long)zh * sW2;
  C += (long)zb * sC1 + (long)zh * sC2;
  if (R) R += (long)zb * sC1 + (long)zh * sC2;

  const int m0   = blockIdx.y * 128;
  const int n0   = blockIdx.x * 128;
  const int tid  = threadIdx.x;
  const int lane = tid & 31;
  const int wave = tid >> 5;
  const int wm   = wave >> 1;        // 0..3 : 32-row slab
  const int wn   = wave & 1;         // 0..1 : 64-col slab
  const int half = lane >> 4;
  const int l16  = lane & 15;

  v8f acc[2][4];
#pragma unroll
  for (int tm = 0; tm < 2; ++tm)
#pragma unroll
    for (int tn = 0; tn < 4; ++tn)
#pragma unroll
      for (int r = 0; r < 8; ++r) acc[tm][tn][r] = 0.0f;

  for (int k0 = 0; k0 < K; k0 += 32) {
    // ---- stage A tile (128 x 32): branch-free (M%128==0, K%32==0, aligned) ----
#pragma unroll
    for (int i = 0; i < 2; ++i) {
      int linear = i * 2048 + tid * 8;
      int row = linear >> 5, col = linear & 31;
      const float* src = A + (long)(m0 + row) * lda + k0 + col;
      float4 f0 = *(const float4*)(src);
      float4 f1 = *(const float4*)(src + 4);
      u64_t* dst = (u64_t*)&lA[row][col];
      dst[0] = pack4bf(f0.x, f0.y, f0.z, f0.w);
      dst[1] = pack4bf(f1.x, f1.y, f1.z, f1.w);
    }

    // ---- stage W tile (32 x 128) ----
    if (!TRANSB) {
#pragma unroll
      for (int i = 0; i < 2; ++i) {
        int linear = i * 2048 + tid * 8;
        int kk = linear >> 7, nn = linear & 127;
        int gn = n0 + nn;
        const float* src = W + (long)(k0 + kk) * ldw + gn;
        u64_t* dst = (u64_t*)&lB[kk][nn];
        if (!GUARDN || (gn + 7 < N)) {
          if (ALW) {
            float4 f0 = *(const float4*)(src);
            float4 f1 = *(const float4*)(src + 4);
            dst[0] = pack4bf(f0.x, f0.y, f0.z, f0.w);
            dst[1] = pack4bf(f1.x, f1.y, f1.z, f1.w);
          } else {
            dst[0] = pack4bf(src[0], src[1], src[2], src[3]);
            dst[1] = pack4bf(src[4], src[5], src[6], src[7]);
          }
        } else {
          float t[8];
#pragma unroll
          for (int e = 0; e < 8; ++e) t[e] = ((gn + e) < N) ? src[e] : 0.f;
          dst[0] = pack4bf(t[0], t[1], t[2], t[3]);
          dst[1] = pack4bf(t[4], t[5], t[6], t[7]);
        }
      }
    } else {
      // W stored [N,K] (scores GEMM): always fully in-bounds in this workload.
#pragma unroll
      for (int i = 0; i < 2; ++i) {
        int linear = i * 2048 + tid * 8;
        int nn = linear >> 5, kk = linear & 31;
        const float* src = W + (long)(n0 + nn) * ldw + k0 + kk;
        float4 f0 = *(const float4*)(src);
        float4 f1 = *(const float4*)(src + 4);
        lB[kk + 0][nn] = f2bf(f0.x); lB[kk + 1][nn] = f2bf(f0.y);
        lB[kk + 2][nn] = f2bf(f0.z); lB[kk + 3][nn] = f2bf(f0.w);
        lB[kk + 4][nn] = f2bf(f1.x); lB[kk + 5][nn] = f2bf(f1.y);
        lB[kk + 6][nn] = f2bf(f1.z); lB[kk + 7][nn] = f2bf(f1.w);
      }
    }

    // ---- prefetch next K tile while we compute this one ----
    if (k0 + 32 < K) {
      __builtin_prefetch(&A[(long)(m0 + (tid >> 1)) * lda + k0 + 32 + (tid & 1) * 16], 0, 0);
      if (!TRANSB) {
        int wn2 = n0 + (tid & 7) * 16;
        if (!GUARDN || wn2 < N)
          __builtin_prefetch(&W[(long)(k0 + 32 + (tid >> 3)) * ldw + wn2], 0, 0);
      }
    }
    __syncthreads();

    // ---- build fragments from LDS ----
    v4u64 af[2], bfr[4];
#pragma unroll
    for (int tm = 0; tm < 2; ++tm) {
      const int row = wm * 32 + tm * 16 + l16;
      const unsigned short* p0 = &lA[row][half * 8];
      const unsigned short* p1 = &lA[row][16 + half * 8];
      af[tm][0] = *(const u64_t*)(p0);
      af[tm][1] = *(const u64_t*)(p0 + 4);
      af[tm][2] = *(const u64_t*)(p1);
      af[tm][3] = *(const u64_t*)(p1 + 4);
    }
#pragma unroll
    for (int tn = 0; tn < 4; ++tn) {
      const int cn = wn * 64 + tn * 16 + l16;
#pragma unroll
      for (int j = 0; j < 4; ++j) {
        const int kb = (j < 2 ? 0 : 16) + half * 8 + (j & 1) * 4;
        bfr[tn][j] = (u64_t)lB[kb][cn] | ((u64_t)lB[kb + 1][cn] << 16) |
                     ((u64_t)lB[kb + 2][cn] << 32) | ((u64_t)lB[kb + 3][cn] << 48);
      }
    }

    // ---- 8 WMMAs per wave per K-step ----
#pragma unroll
    for (int tm = 0; tm < 2; ++tm)
#pragma unroll
      for (int tn = 0; tn < 4; ++tn)
        acc[tm][tn] = __builtin_amdgcn_wmma_f32_16x16x32_bf16(
            false, __builtin_bit_cast(v16bf, af[tm]),
            false, __builtin_bit_cast(v16bf, bfr[tn]),
            (short)0, acc[tm][tn], false, false);

    __syncthreads();
  }

  // ---- epilogue: bias -> gelu -> residual -> store (rows never ragged) ----
#pragma unroll
  for (int tn = 0; tn < 4; ++tn) {
    const int col = n0 + wn * 64 + tn * 16 + l16;
    if (GUARDN && col >= N) continue;
    const float bb = bias ? bias[col] : 0.f;
#pragma unroll
    for (int tm = 0; tm < 2; ++tm) {
#pragma unroll
      for (int r = 0; r < 8; ++r) {
        int row = m0 + wm * 32 + tm * 16 + half * 8 + r;
        float vv = acc[tm][tn][r] + bb;
        if (GELU) vv = gelu_tanh_dev(vv);
        long off = (long)row * ldc + col;
        if (R) vv += R[off];
        C[off] = vv;
      }
    }
  }
}

// ---------------------------------------------------------------------------
#define F_TRANSB 1
#define F_GELU   2

static void launch_gemm(hipStream_t stream,
                        const float* A, int lda, long sA1, long sA2,
                        const float* W, int ldw, long sW1, long sW2,
                        float* C, int ldc, long sC1, long sC2,
                        const float* R, const float* bias,
                        int M, int N, int K, int batches, int hdiv, int flags) {
  dim3 grid((N + 127) / 128, (M + 127) / 128, batches);
  const bool guardN = (N % 128) != 0;
  const bool alW    = (ldw % 4) == 0;
  if (flags & F_TRANSB) {
    gemm_bf16_wmma<true, false, false, true><<<grid, 256, 0, stream>>>(
        A, lda, sA1, sA2, W, ldw, sW1, sW2, C, ldc, sC1, sC2, R, bias, M, N, K, hdiv);
  } else if (guardN && alW) {
    gemm_bf16_wmma<false, false, true, true><<<grid, 256, 0, stream>>>(
        A, lda, sA1, sA2, W, ldw, sW1, sW2, C, ldc, sC1, sC2, R, bias, M, N, K, hdiv);
  } else if (guardN) {
    gemm_bf16_wmma<false, false, true, false><<<grid, 256, 0, stream>>>(
        A, lda, sA1, sA2, W, ldw, sW1, sW2, C, ldc, sC1, sC2, R, bias, M, N, K, hdiv);
  } else if (flags & F_GELU) {
    gemm_bf16_wmma<false, true, false, true><<<grid, 256, 0, stream>>>(
        A, lda, sA1, sA2, W, ldw, sW1, sW2, C, ldc, sC1, sC2, R, bias, M, N, K, hdiv);
  } else {
    gemm_bf16_wmma<false, false, false, true><<<grid, 256, 0, stream>>>(
        A, lda, sA1, sA2, W, ldw, sW1, sW2, C, ldc, sC1, sC2, R, bias, M, N, K, hdiv);
  }
}

extern "C" void kernel_launch(void* const* d_in, const int* in_sizes, int n_in,
                              void* d_out, int out_size, void* d_ws, size_t ws_size,
                              hipStream_t stream) {
  (void)in_sizes; (void)n_in; (void)out_size; (void)ws_size;

  const int Vv = 50257, Dd = 768, Hh = 12, Ll = 12, Bb = 4, Ss = 512, HD = 64;
  const int MS = Bb * Ss;                         // 2048 token rows
  const int D4 = 4 * Dd;                          // 3072

  const int*   xin      = (const int*)  d_in[0];
  const float* tok_emb  = (const float*)d_in[1];
  const float* pos_emb  = (const float*)d_in[2];
  const float* ln1s     = (const float*)d_in[3];
  const float* ln1b     = (const float*)d_in[4];
  const float* wq       = (const float*)d_in[5];
  const float* wk       = (const float*)d_in[6];
  const float* wv       = (const float*)d_in[7];
  const float* wo       = (const float*)d_in[8];
  const float* bo       = (const float*)d_in[9];
  const float* ln2s     = (const float*)d_in[10];
  const float* ln2b     = (const float*)d_in[11];
  const float* w1       = (const float*)d_in[12];
  const float* b1       = (const float*)d_in[13];
  const float* w2       = (const float*)d_in[14];
  const float* b2       = (const float*)d_in[15];
  const float* fins     = (const float*)d_in[16];
  const float* finb     = (const float*)d_in[17];
  const float* w_out    = (const float*)d_in[18];
  float*       out      = (float*)d_out;

  // workspace layout (floats): ~113 MB total
  float* ws  = (float*)d_ws;
  float* h   = ws;
  float* hn  = h   + (size_t)MS * Dd;
  float* q   = hn  + (size_t)MS * Dd;
  float* k   = q   + (size_t)MS * Dd;
  float* v   = k   + (size_t)MS * Dd;
  float* ctx = v   + (size_t)MS * Dd;
  float* ff  = ctx + (size_t)MS * Dd;
  float* sc  = ff  + (size_t)MS * D4;             // B*H*S*S scores

  const long sBH_qkv = (long)Ss * Dd;             // per-b stride in q/k/v/ctx
  const long sB_sc   = (long)Hh * Ss * Ss;        // per-b stride in scores
  const long sH_sc   = (long)Ss * Ss;             // per-h stride in scores

  long total = (long)MS * Dd;
  embed_kernel<<<(unsigned)((total + 255) / 256), 256, 0, stream>>>(
      xin, tok_emb, pos_emb, h, Ss, Dd, total);

  for (int l = 0; l < Ll; ++l) {
    const float* Wq = wq + (long)l * Dd * Dd;
    const float* Wk = wk + (long)l * Dd * Dd;
    const float* Wv = wv + (long)l * Dd * Dd;
    const float* Wo = wo + (long)l * Dd * Dd;
    const float* W1 = w1 + (long)l * Dd * D4;
    const float* W2 = w2 + (long)l * D4 * Dd;

    // hn = LN1(h)
    ln_kernel<<<MS, 256, 0, stream>>>(h, ln1s + l * Dd, ln1b + l * Dd, hn, Dd);

    // q/k/v = hn @ Wq/Wk/Wv
    launch_gemm(stream, hn, Dd, 0, 0, Wq, Dd, 0, 0, q, Dd, 0, 0,
                nullptr, nullptr, MS, Dd, Dd, 1, 1, 0);
    launch_gemm(stream, hn, Dd, 0, 0, Wk, Dd, 0, 0, k, Dd, 0, 0,
                nullptr, nullptr, MS, Dd, Dd, 1, 1, 0);
    launch_gemm(stream, hn, Dd, 0, 0, Wv, Dd, 0, 0, v, Dd, 0, 0,
                nullptr, nullptr, MS, Dd, Dd, 1, 1, 0);

    // scores[b,h] = Q[b,:,h,:] @ K[b,:,h,:]^T   (batched over 48, transB)
    launch_gemm(stream, q, Dd, sBH_qkv, HD, k, Dd, sBH_qkv, HD,
                sc, Ss, sB_sc, sH_sc, nullptr, nullptr,
                Ss, Ss, HD, Bb * Hh, Hh, F_TRANSB);

    // softmax(scale * scores) with causal mask
    softmax_causal<<<Bb * Hh * Ss, 256, 0, stream>>>(sc, Ss, 0.125f);

    // ctx[b,:,h,:] = attn[b,h] @ V[b,:,h,:]
    launch_gemm(stream, sc, Ss, sB_sc, sH_sc, v, Dd, sBH_qkv, HD,
                ctx, Dd, sBH_qkv, HD, nullptr, nullptr,
                Ss, HD, Ss, Bb * Hh, Hh, 0);

    // h = h + ctx @ Wo + Bo
    launch_gemm(stream, ctx, Dd, 0, 0, Wo, Dd, 0, 0, h, Dd, 0, 0,
                h, bo + l * Dd, MS, Dd, Dd, 1, 1, 0);

    // hn = LN2(h)
    ln_kernel<<<MS, 256, 0, stream>>>(h, ln2s + l * Dd, ln2b + l * Dd, hn, Dd);

    // ff = gelu(hn @ W1 + b1)
    launch_gemm(stream, hn, Dd, 0, 0, W1, D4, 0, 0, ff, D4, 0, 0,
                nullptr, b1 + (long)l * D4, MS, D4, Dd, 1, 1, F_GELU);

    // h = h + ff @ W2 + b2
    launch_gemm(stream, ff, D4, 0, 0, W2, Dd, 0, 0, h, Dd, 0, 0,
                h, b2 + l * Dd, MS, Dd, D4, 1, 1, 0);
  }

  // final LN + logits
  ln_kernel<<<MS, 256, 0, stream>>>(h, fins, finb, hn, Dd);
  launch_gemm(stream, hn, Dd, 0, 0, w_out, Vv, 0, 0, out, Vv, 0, 0,
              nullptr, nullptr, MS, Vv, Dd, 1, 1, 0);
}
